// Model_3427383902569
// MI455X (gfx1250) — matrix operand order repaired and checked by reading, not compiled, not run on hardware
//
#include <hip/hip_runtime.h>
#include <hip/hip_bf16.h>

// ---------------------------------------------------------------------------
// CDNA5 (gfx1250) sparse 3D U-Net forward.
// - All channel contractions on v_wmma_f32_16x16x32_bf16 (wave32 tile:
//   16 voxels x 16 cout, K = taps * Cin in chunks of 32).
// - Activations stored bf16 with zero-padded x-halo -> unconditional A loads.
// - Per-block weight tile staged to LDS via TDM tensor_load_to_lds (6-arg
//   builtin on this toolchain); B fragments served by ds_load_b128.
// ---------------------------------------------------------------------------

typedef __bf16 bf16_t;
typedef bf16_t bf16x8  __attribute__((ext_vector_type(8)));
typedef bf16_t bf16x16 __attribute__((ext_vector_type(16)));
typedef float  f32x8   __attribute__((ext_vector_type(8)));

#define PADL 16
#define PADR 32
#define XPAD (PADL + PADR)

__device__ __forceinline__ bf16x16 load_frag(const bf16_t* lo_p, const bf16_t* hi_p) {
    bf16x8 lo = *(const bf16x8*)lo_p;
    bf16x8 hi = *(const bf16x8*)hi_p;
    return __builtin_shufflevector(lo, hi, 0,1,2,3,4,5,6,7,8,9,10,11,12,13,14,15);
}

__device__ __forceinline__ f32x8 wmma_bf16(bf16x16 a, bf16x16 b, f32x8 c) {
    return __builtin_amdgcn_wmma_f32_16x16x32_bf16(
        /*neg_a=*/false, a, /*neg_b=*/false, b,
        /*c_mod=*/(short)0, c, /*reuse_a=*/false, /*reuse_b=*/false);
}

// ---------------------------------------------------------------------------
// Stage n16 * 16 bytes of packed weights from global into LDS (block-wide).
// Preferred path: single Tensor Data Mover 1-D descriptor issued by wave 0,
// completion via s_wait_tensorcnt. Fallback: cooperative vec4 copy.
// ---------------------------------------------------------------------------
__device__ __forceinline__ void stage_to_lds(const bf16_t* __restrict__ g,
                                             bf16_t* __restrict__ l, int n16) {
#if __has_builtin(__builtin_amdgcn_tensor_load_to_lds) && __has_builtin(__builtin_amdgcn_s_wait_tensorcnt)
    if (threadIdx.x < 32) {
        typedef unsigned int u32x4 __attribute__((ext_vector_type(4)));
        typedef int          i32x8 __attribute__((ext_vector_type(8)));
        typedef int          i32x4 __attribute__((ext_vector_type(4)));
        unsigned long long ga = (unsigned long long)(uintptr_t)g;
        unsigned int       la = (unsigned int)(uintptr_t)l;     // LDS byte address
        unsigned int      ndw = (unsigned int)n16 * 4u;         // tile in DWORDs
        u32x4 g0;
        g0[0] = 1u;                                              // count=1 (valid user D#)
        g0[1] = la;                                              // lds_addr
        g0[2] = (unsigned int)(ga & 0xffffffffu);                // global_addr[31:0]
        g0[3] = (unsigned int)((ga >> 32) & 0x01ffffffu) | 0x80000000u; // addr[56:32] | type=2
        i32x8 g1;
        g1[0] = 0x00020000;                                      // data_size = 4 bytes
        g1[1] = (int)((ndw & 0xffffu) << 16);                    // tensor_dim0[15:0]
        g1[2] = (int)((ndw >> 16) & 0xffffu) | (1 << 16);        // tensor_dim0[31:16] | tensor_dim1=1
        g1[3] = (int)((ndw & 0xffffu) << 16);                    // tile_dim0 = ndw
        g1[4] = 0;                                               // tile_dim1/2 unused
        g1[5] = (int)ndw;                                        // tensor_dim0_stride lo
        g1[6] = 0;
        g1[7] = 0;
        i32x4 z4 = {};
        i32x8 z8 = {};
        __builtin_amdgcn_tensor_load_to_lds(g0, g1, z4, z4, z8, 0);
        __builtin_amdgcn_s_wait_tensorcnt(0);
    }
#else
    const uint4* s = (const uint4*)g;
    uint4*       d = (uint4*)l;
    for (int i = threadIdx.x; i < n16; i += blockDim.x) d[i] = s[i];
#endif
    __syncthreads();
}

// ---------------------------------------------------------------------------
// Weight packing: DHWIO f32 -> bf16 B-fragment layout, nb-major so each
// cout-tile's full weight set is one contiguous blob for the TDM:
// pk[((nb*T*KC + t*KC+kc)*32 + lane)*16 + v]
//   = w[t][kc*32 + (lane>=16?16:0) + v][nb*16 + (lane&15)]
// ---------------------------------------------------------------------------
__global__ void pack_weights(const float* __restrict__ w, bf16_t* __restrict__ pk,
                             int T, int Ci, int Co) {
    int KC = Ci >> 5, NB = Co >> 4;
    int total = T * KC * NB * 512;
    int i = blockIdx.x * blockDim.x + threadIdx.x;
    if (i >= total) return;
    int v    = i & 15;
    int lane = (i >> 4) & 31;
    int frag = i >> 9;
    int f  = frag % (T * KC);
    int nb = frag / (T * KC);
    int t  = f / KC, kc = f % KC;
    int ci = kc * 32 + ((lane & 16) ? 16 : 0) + v;
    int co = nb * 16 + (lane & 15);
    pk[i] = (bf16_t)w[((size_t)t * Ci + ci) * Co + co];
}

// ---------------------------------------------------------------------------
// Scatter + mask pooling
// ---------------------------------------------------------------------------
__global__ void scatter_kernel(const int* __restrict__ coords, const float* __restrict__ feats,
                               float* __restrict__ grid, float* __restrict__ mask, int N) {
    int n = blockIdx.x * blockDim.x + threadIdx.x;
    if (n >= N) return;
    int i = coords[3 * n + 0], j = coords[3 * n + 1], k = coords[3 * n + 2];
    int vox = (i * 100 + j) * 100 + k;
    atomicAdd(&grid[vox], feats[n]);
    mask[vox] = 1.0f;
}

__global__ void pool_mask(const float* __restrict__ in, float* __restrict__ out,
                          int Do, int Ho, int Wo) {
    int v = blockIdx.x * blockDim.x + threadIdx.x;
    if (v >= Do * Ho * Wo) return;
    int w = v % Wo, h = (v / Wo) % Ho, d = v / (Wo * Ho);
    int Wi = Wo * 2, Hi = Ho * 2;
    float m = 0.0f;
    for (int a = 0; a < 2; ++a)
        for (int b = 0; b < 2; ++b)
            for (int c = 0; c < 2; ++c)
                m = fmaxf(m, in[((size_t)(2 * d + a) * Hi + (2 * h + b)) * Wi + (2 * w + c)]);
    out[v] = m;
}

// ---------------------------------------------------------------------------
// Input conv: 1 -> 32 channels, 3^3 SAME, * mask (K=1 per tap: VALU).
// ---------------------------------------------------------------------------
__global__ void conv_in_kernel(const float* __restrict__ grid, const float* __restrict__ w,
                               const float* __restrict__ mask, float* __restrict__ out,
                               int D, int H, int W) {
    int v = blockIdx.x * blockDim.x + threadIdx.x;
    if (v >= D * H * W) return;
    int ww = v % W, h = (v / W) % H, d = v / (W * H);
    float acc[32];
#pragma unroll
    for (int c = 0; c < 32; ++c) acc[c] = 0.0f;
    for (int td = -1; td <= 1; ++td) {
        int dd = d + td; if ((unsigned)dd >= (unsigned)D) continue;
        for (int th = -1; th <= 1; ++th) {
            int hh = h + th; if ((unsigned)hh >= (unsigned)H) continue;
            for (int tw = -1; tw <= 1; ++tw) {
                int wv = ww + tw; if ((unsigned)wv >= (unsigned)W) continue;
                float g = grid[((size_t)dd * H + hh) * W + wv];
                int t = (td + 1) * 9 + (th + 1) * 3 + (tw + 1);
#pragma unroll
                for (int c = 0; c < 32; ++c) acc[c] += g * w[t * 32 + c];
            }
        }
    }
    float mk = mask[v];
#pragma unroll
    for (int c = 0; c < 32; ++c) out[(size_t)v * 32 + c] = acc[c] * mk;
}

// ---------------------------------------------------------------------------
// BN + ReLU + mask, f32 (unpadded) -> bf16 (x-padded), channel offset for
// fused concat.  bn layout: stacked (4, Cbn): gamma, beta, mu, var.
// ---------------------------------------------------------------------------
__global__ void bnrelu_kernel(const float* __restrict__ src, int Csrc,
                              bf16_t* __restrict__ dst, int Cdst, int dstOff,
                              const float* __restrict__ bn, int Cbn, int bnOff,
                              const float* __restrict__ mask, int nvox, int W) {
    int i = blockIdx.x * blockDim.x + threadIdx.x;
    if (i >= nvox * Csrc) return;
    int c = i % Csrc;
    int v = i / Csrc;
    int x = v % W, row = v / W;
    float g  = bn[bnOff + c];
    float b  = bn[Cbn + bnOff + c];
    float mu = bn[2 * Cbn + bnOff + c];
    float vr = bn[3 * Cbn + bnOff + c];
    float val = (src[i] - mu) * (g * rsqrtf(vr + 1e-4f)) + b;
    val = fmaxf(val, 0.0f) * mask[v];
    size_t pv = (size_t)row * (W + XPAD) + PADL + x;
    dst[pv * Cdst + dstOff + c] = (bf16_t)val;
}

// ---------------------------------------------------------------------------
// 3x3x3 SAME conv, WMMA. blockIdx.y = cout tile; weights staged in LDS.
// ---------------------------------------------------------------------------
template <int KC, int NB>
__global__ void conv3_wmma(const bf16_t* __restrict__ act, const bf16_t* __restrict__ wpk,
                           const float* __restrict__ mask, float* __restrict__ out,
                           int D, int H, int W) {
    constexpr int T = 27;
    constexpr int Cin = KC * 32, Cout = NB * 16;
    extern __shared__ char smem_raw[];
    bf16_t* sB = (bf16_t*)smem_raw;

    const int nb = blockIdx.y;
    stage_to_lds(wpk + (size_t)nb * (T * KC * 512), sB, T * KC * 64);

    const int WT = (W + 15) >> 4;
    const int plane = D * H * WT;
    const int wid = blockIdx.x * (blockDim.x >> 5) + (threadIdx.x >> 5);
    if (wid >= plane) return;
    const int lane = threadIdx.x & 31;
    const int Wp = W + XPAD;

    int id = wid;
    int wt = id % WT; id /= WT;
    int h  = id % H;  int d = id / H;

    const int m     = lane & 15;
    const int halfA = (lane & 16) ? 16 : 0;
    const int w0    = wt << 4;
    const bf16_t* sBl = sB + lane * 16;

    f32x8 acc = {};
    for (int td = -1; td <= 1; ++td) {
        int dd = d + td; if ((unsigned)dd >= (unsigned)D) continue;
        for (int th = -1; th <= 1; ++th) {
            int hh = h + th; if ((unsigned)hh >= (unsigned)H) continue;
            const bf16_t* arow =
                act + (((size_t)(dd * H + hh)) * Wp + (PADL - 1 + w0 + m)) * Cin + halfA;
            const int f0 = ((td + 1) * 9 + (th + 1) * 3) * KC;
#pragma unroll
            for (int tw = 0; tw < 3; ++tw) {
                bf16x16 Af[KC], Bf[KC];
#pragma unroll
                for (int kc = 0; kc < KC; ++kc) {
                    const bf16_t* ap = arow + tw * Cin + kc * 32;
                    Af[kc] = load_frag(ap, ap + 8);
                    const bf16_t* bq = sBl + (size_t)(f0 + tw * KC + kc) * 512;
                    Bf[kc] = load_frag(bq, bq + 8);
                }
#pragma unroll
                for (int kc = 0; kc < KC; ++kc) acc = wmma_bf16(Af[kc], Bf[kc], acc);
            }
        }
    }
    const int n = lane & 15;
    const int mrow0 = (lane & 16) ? 8 : 0;
    const size_t orow = ((size_t)d * H + h) * W;
#pragma unroll
    for (int r = 0; r < 8; ++r) {
        int w = w0 + mrow0 + r;
        if (w < W) {
            size_t vox = orow + w;
            out[vox * Cout + nb * 16 + n] = acc[r] * mask[vox];
        }
    }
}

// ---------------------------------------------------------------------------
// 2x2x2 stride-2 VALID down-conv, WMMA, no mask.
// ---------------------------------------------------------------------------
template <int KC, int NB>
__global__ void convdown_wmma(const bf16_t* __restrict__ act, const bf16_t* __restrict__ wpk,
                              float* __restrict__ out, int Do, int Ho, int Wo) {
    constexpr int T = 8;
    constexpr int Cin = KC * 32, Cout = NB * 16;
    extern __shared__ char smem_raw[];
    bf16_t* sB = (bf16_t*)smem_raw;

    const int nb = blockIdx.y;
    stage_to_lds(wpk + (size_t)nb * (T * KC * 512), sB, T * KC * 64);

    const int WT = (Wo + 15) >> 4;
    const int plane = Do * Ho * WT;
    const int wid = blockIdx.x * (blockDim.x >> 5) + (threadIdx.x >> 5);
    if (wid >= plane) return;
    const int lane = threadIdx.x & 31;

    int id = wid;
    int wt = id % WT; id /= WT;
    int h  = id % Ho; int d = id / Ho;

    const int Hi = Ho * 2, Wpi = Wo * 2 + XPAD;
    const int m     = lane & 15;
    const int halfA = (lane & 16) ? 16 : 0;
    const int w0    = wt << 4;
    const int win0  = PADL + 2 * (w0 + m);
    const bf16_t* sBl = sB + lane * 16;

    f32x8 acc = {};
#pragma unroll
    for (int td = 0; td < 2; ++td) {
#pragma unroll
        for (int th = 0; th < 2; ++th) {
            const bf16_t* arow =
                act + (((size_t)(2 * d + td) * Hi + (2 * h + th)) * Wpi + win0) * Cin + halfA;
            const int f0 = (td * 4 + th * 2) * KC;
#pragma unroll
            for (int tw = 0; tw < 2; ++tw) {
                bf16x16 Af[KC], Bf[KC];
#pragma unroll
                for (int kc = 0; kc < KC; ++kc) {
                    const bf16_t* ap = arow + tw * Cin + kc * 32;
                    Af[kc] = load_frag(ap, ap + 8);
                    const bf16_t* bq = sBl + (size_t)(f0 + tw * KC + kc) * 512;
                    Bf[kc] = load_frag(bq, bq + 8);
                }
#pragma unroll
                for (int kc = 0; kc < KC; ++kc) acc = wmma_bf16(Af[kc], Bf[kc], acc);
            }
        }
    }
    const int n = lane & 15;
    const int mrow0 = (lane & 16) ? 8 : 0;
    const size_t orow = ((size_t)d * Ho + h) * Wo;
#pragma unroll
    for (int r = 0; r < 8; ++r) {
        int w = w0 + mrow0 + r;
        if (w < Wo) out[(orow + w) * Cout + nb * 16 + n] = acc[r];
    }
}

// ---------------------------------------------------------------------------
// 2x2x2 stride-2 transpose conv (VALID), WMMA, * mask.
// out[d,h,w] = sum_ci in[d>>1,h>>1,w>>1,ci] * w[1-(d&1),1-(h&1),1-(w&1),ci,co]
// Tiles grouped by output x-parity -> tap (B) is wave-uniform.
// ---------------------------------------------------------------------------
template <int KC, int NB>
__global__ void convup_wmma(const bf16_t* __restrict__ act, const bf16_t* __restrict__ wpk,
                            const float* __restrict__ mask, float* __restrict__ out,
                            int Do, int Ho, int Wo) {
    constexpr int T = 8;
    constexpr int Cin = KC * 32, Cout = NB * 16;
    extern __shared__ char smem_raw[];
    bf16_t* sB = (bf16_t*)smem_raw;

    const int nb = blockIdx.y;
    stage_to_lds(wpk + (size_t)nb * (T * KC * 512), sB, T * KC * 64);

    const int Wq = Wo >> 1;
    const int WT = (Wq + 15) >> 4;
    const int plane = Do * Ho * 2 * WT;
    const int wid = blockIdx.x * (blockDim.x >> 5) + (threadIdx.x >> 5);
    if (wid >= plane) return;
    const int lane = threadIdx.x & 31;

    int id = wid;
    int wt = id % WT; id /= WT;
    int p  = id & 1;  id >>= 1;
    int h  = id % Ho; int d = id / Ho;

    const int Hi = Ho >> 1, Wpi = (Wo >> 1) + XPAD;
    const int m     = lane & 15;
    const int halfA = (lane & 16) ? 16 : 0;
    const int q0    = wt << 4;
    const bf16_t* arow =
        act + (((size_t)(d >> 1) * Hi + (h >> 1)) * Wpi + PADL + q0 + m) * Cin + halfA;
    const int t = (1 - (d & 1)) * 4 + (1 - (h & 1)) * 2 + (1 - p);
    const bf16_t* sBl = sB + lane * 16;

    f32x8 acc = {};
    bf16x16 Af[KC], Bf[KC];
#pragma unroll
    for (int kc = 0; kc < KC; ++kc) {
        const bf16_t* ap = arow + kc * 32;
        Af[kc] = load_frag(ap, ap + 8);
        const bf16_t* bq = sBl + (size_t)(t * KC + kc) * 512;
        Bf[kc] = load_frag(bq, bq + 8);
    }
#pragma unroll
    for (int kc = 0; kc < KC; ++kc) acc = wmma_bf16(Af[kc], Bf[kc], acc);

    const int n = lane & 15;
    const int mrow0 = (lane & 16) ? 8 : 0;
    const size_t orow = ((size_t)d * Ho + h) * Wo;
#pragma unroll
    for (int r = 0; r < 8; ++r) {
        int qr = q0 + mrow0 + r;
        if (qr < Wq) {
            size_t vox = orow + (p + 2 * qr);
            out[vox * Cout + nb * 16 + n] = acc[r] * mask[vox];
        }
    }
}

// ---------------------------------------------------------------------------
// Head: bnrelu(y0, bnJ) * mask, gather, @ linW (32x3) + linb.
// ---------------------------------------------------------------------------
__global__ void head_kernel(const int* __restrict__ coords, const float* __restrict__ y0,
                            const float* __restrict__ bnJ, const float* __restrict__ mask,
                            const float* __restrict__ linW, const float* __restrict__ linb,
                            float* __restrict__ out, int N) {
    int n = blockIdx.x * blockDim.x + threadIdx.x;
    if (n >= N) return;
    int i = coords[3 * n + 0], j = coords[3 * n + 1], k = coords[3 * n + 2];
    size_t vox = ((size_t)i * 100 + j) * 100 + k;
    float mk = mask[vox];
    float a0 = linb[0], a1 = linb[1], a2 = linb[2];
#pragma unroll
    for (int c = 0; c < 32; ++c) {
        float g  = bnJ[c], b = bnJ[32 + c], mu = bnJ[64 + c], vr = bnJ[96 + c];
        float val = (y0[vox * 32 + c] - mu) * (g * rsqrtf(vr + 1e-4f)) + b;
        val = fmaxf(val, 0.0f) * mk;
        a0 += val * linW[c * 3 + 0];
        a1 += val * linW[c * 3 + 1];
        a2 += val * linW[c * 3 + 2];
    }
    out[3 * n + 0] = a0; out[3 * n + 1] = a1; out[3 * n + 2] = a2;
}

// ---------------------------------------------------------------------------
// Host orchestration
// ---------------------------------------------------------------------------
static inline int cdiv_i(int a, int b) { return (a + b - 1) / b; }

extern "C" void kernel_launch(void* const* d_in, const int* in_sizes, int n_in,
                              void* d_out, int out_size, void* d_ws, size_t ws_size,
                              hipStream_t stream) {
    (void)n_in; (void)out_size; (void)ws_size;
    const int*   coords  = (const int*)  d_in[0];
    const float* feats   = (const float*)d_in[1];
    const float* w_in    = (const float*)d_in[2];
    const float* w0a     = (const float*)d_in[3];
    const float* wdown0  = (const float*)d_in[4];
    const float* w1a     = (const float*)d_in[5];
    const float* wdown1  = (const float*)d_in[6];
    const float* w2      = (const float*)d_in[7];
    const float* wup1    = (const float*)d_in[8];
    const float* w1post  = (const float*)d_in[9];
    const float* wup0    = (const float*)d_in[10];
    const float* w0post  = (const float*)d_in[11];
    const float* bnA = (const float*)d_in[12];
    const float* bnB = (const float*)d_in[13];
    const float* bnC = (const float*)d_in[14];
    const float* bnD = (const float*)d_in[15];
    const float* bnE = (const float*)d_in[16];
    const float* bnF = (const float*)d_in[17];
    const float* bnG = (const float*)d_in[18];
    const float* bnH = (const float*)d_in[19];
    const float* bnI = (const float*)d_in[20];
    const float* bnJ = (const float*)d_in[21];
    const float* linW = (const float*)d_in[22];
    const float* linb = (const float*)d_in[23];
    const int N = in_sizes[0] / 3;

    constexpr size_t V0 = 1000000, V1 = 125000, V2 = 15625;
    constexpr size_t R0 = 100 * 100, R1 = 50 * 50, R2 = 25 * 25;   // rows (d*h)
    constexpr size_t P0 = R0 * (100 + XPAD);                        // padded voxels
    constexpr size_t P1 = R1 * (50 + XPAD);
    constexpr size_t P2 = R2 * (25 + XPAD);

    size_t off = 0;
    auto take = [&](size_t bytes) -> char* {
        char* p = (char*)d_ws + off;
        off += (bytes + 255) & ~(size_t)255;
        return p;
    };
    float*  grid  = (float*) take(V0 * 4);
    float*  mask0 = (float*) take(V0 * 4);
    float*  mask1 = (float*) take(V1 * 4);
    float*  mask2 = (float*) take(V2 * 4);
    float*  x     = (float*) take(V0 * 32 * 4);   // also u0
    bf16_t* actA  = (bf16_t*)take(P0 * 32 * 2);   // also actB
    float*  x0    = (float*) take(V0 * 32 * 4);   // also y0
    float*  d0    = (float*) take(V1 * 64 * 4);   // also y1
    bf16_t* actC  = (bf16_t*)take(P1 * 64 * 2);   // also actH
    float*  x1    = (float*) take(V1 * 64 * 4);
    bf16_t* actD  = (bf16_t*)take(P1 * 64 * 2);
    float*  d1    = (float*) take(V2 * 96 * 4);
    bf16_t* actE  = (bf16_t*)take(P2 * 96 * 2);
    float*  x2    = (float*) take(V2 * 96 * 4);
    bf16_t* actF  = (bf16_t*)take(P2 * 96 * 2);
    float*  u1    = (float*) take(V1 * 64 * 4);
    bf16_t* actG  = (bf16_t*)take(P1 * 128 * 2);
    bf16_t* actI  = (bf16_t*)take(P0 * 64 * 2);
    bf16_t* pkW0a    = (bf16_t*)take((size_t)27 * 32 * 32  * 2);
    bf16_t* pkDown0  = (bf16_t*)take((size_t)8  * 32 * 64  * 2);
    bf16_t* pkW1a    = (bf16_t*)take((size_t)27 * 64 * 64  * 2);
    bf16_t* pkDown1  = (bf16_t*)take((size_t)8  * 64 * 96  * 2);
    bf16_t* pkW2     = (bf16_t*)take((size_t)27 * 96 * 96  * 2);
    bf16_t* pkUp1    = (bf16_t*)take((size_t)8  * 96 * 64  * 2);
    bf16_t* pkW1post = (bf16_t*)take((size_t)27 * 128 * 64 * 2);
    bf16_t* pkUp0    = (bf16_t*)take((size_t)8  * 64 * 32  * 2);
    bf16_t* pkW0post = (bf16_t*)take((size_t)27 * 64 * 32  * 2);
    bf16_t* actB = actA;
    float*  y1   = d0;
    bf16_t* actH = actC;
    float*  u0   = x;
    float*  y0   = x0;

    const int TB = 256;

    // zero grid/mask + activation pads (pads stay zero; bnrelu fills interiors)
    (void)hipMemsetAsync(grid,  0, V0 * 4, stream);
    (void)hipMemsetAsync(mask0, 0, V0 * 4, stream);
    (void)hipMemsetAsync(actA, 0, P0 * 32 * 2, stream);
    (void)hipMemsetAsync(actC, 0, P1 * 64 * 2, stream);
    (void)hipMemsetAsync(actD, 0, P1 * 64 * 2, stream);
    (void)hipMemsetAsync(actE, 0, P2 * 96 * 2, stream);
    (void)hipMemsetAsync(actF, 0, P2 * 96 * 2, stream);
    (void)hipMemsetAsync(actG, 0, P1 * 128 * 2, stream);
    (void)hipMemsetAsync(actI, 0, P0 * 64 * 2, stream);

    scatter_kernel<<<cdiv_i(N, TB), TB, 0, stream>>>(coords, feats, grid, mask0, N);
    pool_mask<<<cdiv_i((int)V1, TB), TB, 0, stream>>>(mask0, mask1, 50, 50, 50);
    pool_mask<<<cdiv_i((int)V2, TB), TB, 0, stream>>>(mask1, mask2, 25, 25, 25);

    pack_weights<<<cdiv_i(27 * 32 * 32,  TB), TB, 0, stream>>>(w0a,    pkW0a,    27, 32, 32);
    pack_weights<<<cdiv_i(8  * 32 * 64,  TB), TB, 0, stream>>>(wdown0, pkDown0,  8,  32, 64);
    pack_weights<<<cdiv_i(27 * 64 * 64,  TB), TB, 0, stream>>>(w1a,    pkW1a,    27, 64, 64);
    pack_weights<<<cdiv_i(8  * 64 * 96,  TB), TB, 0, stream>>>(wdown1, pkDown1,  8,  64, 96);
    pack_weights<<<cdiv_i(27 * 96 * 96,  TB), TB, 0, stream>>>(w2,     pkW2,     27, 96, 96);
    pack_weights<<<cdiv_i(8  * 96 * 64,  TB), TB, 0, stream>>>(wup1,   pkUp1,    8,  96, 64);
    pack_weights<<<cdiv_i(27 * 128 * 64, TB), TB, 0, stream>>>(w1post, pkW1post, 27, 128, 64);
    pack_weights<<<cdiv_i(8  * 64 * 32,  TB), TB, 0, stream>>>(wup0,   pkUp0,    8,  64, 32);
    pack_weights<<<cdiv_i(27 * 64 * 32,  TB), TB, 0, stream>>>(w0post, pkW0post, 27, 64, 32);

    // --- encoder ---
    conv_in_kernel<<<cdiv_i((int)V0, TB), TB, 0, stream>>>(grid, w_in, mask0, x, 100, 100, 100);
    bnrelu_kernel<<<cdiv_i((int)V0 * 32, TB), TB, 0, stream>>>(x, 32, actA, 32, 0, bnA, 32, 0, mask0, (int)V0, 100);
    conv3_wmma<1, 2><<<dim3(cdiv_i(100 * 100 * 7, 8), 2), TB, 27 * 1 * 1024, stream>>>(actA, pkW0a, mask0, x0, 100, 100, 100);
    bnrelu_kernel<<<cdiv_i((int)V0 * 32, TB), TB, 0, stream>>>(x0, 32, actB, 32, 0, bnB, 32, 0, mask0, (int)V0, 100);
    convdown_wmma<1, 4><<<dim3(cdiv_i(50 * 50 * 4, 8), 4), TB, 8 * 1 * 1024, stream>>>(actB, pkDown0, d0, 50, 50, 50);
    bnrelu_kernel<<<cdiv_i((int)V1 * 64, TB), TB, 0, stream>>>(d0, 64, actC, 64, 0, bnC, 64, 0, mask1, (int)V1, 50);
    conv3_wmma<2, 4><<<dim3(cdiv_i(50 * 50 * 4, 8), 4), TB, 27 * 2 * 1024, stream>>>(actC, pkW1a, mask1, x1, 50, 50, 50);
    bnrelu_kernel<<<cdiv_i((int)V1 * 64, TB), TB, 0, stream>>>(x1, 64, actD, 64, 0, bnD, 64, 0, mask1, (int)V1, 50);
    convdown_wmma<2, 6><<<dim3(cdiv_i(25 * 25 * 2, 8), 6), TB, 8 * 2 * 1024, stream>>>(actD, pkDown1, d1, 25, 25, 25);
    bnrelu_kernel<<<cdiv_i((int)V2 * 96, TB), TB, 0, stream>>>(d1, 96, actE, 96, 0, bnE, 96, 0, mask2, (int)V2, 25);
    conv3_wmma<3, 6><<<dim3(cdiv_i(25 * 25 * 2, 8), 6), TB, 27 * 3 * 1024, stream>>>(actE, pkW2, mask2, x2, 25, 25, 25);

    // --- decoder level 1 ---
    bnrelu_kernel<<<cdiv_i((int)V2 * 96, TB), TB, 0, stream>>>(x2, 96, actF, 96, 0, bnF, 96, 0, mask2, (int)V2, 25);
    convup_wmma<3, 4><<<dim3(cdiv_i(50 * 50 * 2 * 2, 8), 4), TB, 8 * 3 * 1024, stream>>>(actF, pkUp1, mask1, u1, 50, 50, 50);
    bnrelu_kernel<<<cdiv_i((int)V1 * 64, TB), TB, 0, stream>>>(x1, 64, actG, 128, 0,  bnG, 128, 0,  mask1, (int)V1, 50);
    bnrelu_kernel<<<cdiv_i((int)V1 * 64, TB), TB, 0, stream>>>(u1, 64, actG, 128, 64, bnG, 128, 64, mask1, (int)V1, 50);
    conv3_wmma<4, 4><<<dim3(cdiv_i(50 * 50 * 4, 8), 4), TB, 27 * 4 * 1024, stream>>>(actG, pkW1post, mask1, y1, 50, 50, 50);

    // --- decoder level 0 ---
    bnrelu_kernel<<<cdiv_i((int)V1 * 64, TB), TB, 0, stream>>>(y1, 64, actH, 64, 0, bnH, 64, 0, mask1, (int)V1, 50);
    convup_wmma<2, 2><<<dim3(cdiv_i(100 * 100 * 2 * 4, 8), 2), TB, 8 * 2 * 1024, stream>>>(actH, pkUp0, mask0, u0, 100, 100, 100);
    bnrelu_kernel<<<cdiv_i((int)V0 * 32, TB), TB, 0, stream>>>(x0, 32, actI, 64, 0,  bnI, 64, 0,  mask0, (int)V0, 100);
    bnrelu_kernel<<<cdiv_i((int)V0 * 32, TB), TB, 0, stream>>>(u0, 32, actI, 64, 32, bnI, 64, 32, mask0, (int)V0, 100);
    conv3_wmma<2, 2><<<dim3(cdiv_i(100 * 100 * 7, 8), 2), TB, 27 * 2 * 1024, stream>>>(actI, pkW0post, mask0, y0, 100, 100, 100);

    // --- head ---
    head_kernel<<<cdiv_i(N, TB), TB, 0, stream>>>(coords, y0, bnJ, mask0, linW, linb,
                                                  (float*)d_out, N);
}
